// Net_63299228009240
// MI455X (gfx1250) — matrix operand order repaired and checked by reading, not compiled, not run on hardware
//
#include <hip/hip_runtime.h>
#include <hip/hip_bf16.h>

// ---------------------------------------------------------------------------
// Types for CDNA5 WMMA
// ---------------------------------------------------------------------------
typedef __attribute__((ext_vector_type(16))) __bf16 v16bf;
typedef __attribute__((ext_vector_type(8)))  float  v8f;

union ABFrag {
    unsigned int u[8];
    v16bf        v;
};

// float -> bf16 (round to nearest even-ish)
static __device__ __forceinline__ unsigned short f2bf(float f) {
    unsigned int u = __float_as_uint(f);
    u += 0x7FFFu + ((u >> 16) & 1u);
    return (unsigned short)(u >> 16);
}
static __device__ __forceinline__ float bf2f(unsigned short h) {
    return __uint_as_float(((unsigned int)h) << 16);
}
static __device__ __forceinline__ unsigned int pack2(unsigned short lo, unsigned short hi) {
    return (unsigned int)lo | ((unsigned int)hi << 16);
}

// ---------------------------------------------------------------------------
// CDNA5 async global->LDS copy (16B per lane), tracked by ASYNCcnt.
// VDST VGPR carries the LDS byte offset: low 32 bits of a generic pointer to
// LDS are the LDS-relative address (ISA aperture mapping).
// ---------------------------------------------------------------------------
static __device__ __forceinline__ void async_copy_b128(const void* gptr, void* lptr) {
    unsigned lds = (unsigned)(size_t)lptr;
    asm volatile("global_load_async_to_lds_b128 %0, %1, off"
                 :: "v"(lds), "v"(gptr)
                 : "memory");
}
static __device__ __forceinline__ void wait_async0() {
    asm volatile("s_wait_asynccnt 0x0" ::: "memory");
}

// A-operand fragment: 16x32 bf16, row-major source [row][k], row_stride in elements.
// Per ISA: lanes 0-15 => M rows, K 0..7 (v0..3) & 16..23 (v4..7); lanes 16-31 same rows,
// K 8..15 & 24..31.  So per lane: 8 contiguous bf16 at k0+half*8 and at k0+16+half*8.
static __device__ __forceinline__ v16bf load_a_frag(const unsigned short* base, int row_stride,
                                                    int m0, int k0, int lane) {
    const int r = lane & 15, half = lane >> 4;
    const unsigned short* p = base + (size_t)(m0 + r) * row_stride + k0 + half * 8;
    ABFrag f;
    uint4 a = *(const uint4*)(p);
    uint4 b = *(const uint4*)(p + 16);
    f.u[0] = a.x; f.u[1] = a.y; f.u[2] = a.z; f.u[3] = a.w;
    f.u[4] = b.x; f.u[5] = b.y; f.u[6] = b.z; f.u[7] = b.w;
    return f.v;
}

// B-operand fragment: 32x16 bf16 from *transposed* storage Bt[col][k], stride in elements.
// Per ISA: lane = column; lanes 0-15 K 0..7 & 16..23, lanes 16-31 K 8..15 & 24..31 (same as A)
// => per lane 8 contiguous bf16 at Bt[col][k0 + 8*half] and at Bt[col][k0 + 16 + 8*half].
static __device__ __forceinline__ v16bf load_b_frag(const unsigned short* baseT, int row_stride,
                                                    int n0, int k0, int lane) {
    const int c = lane & 15, half = lane >> 4;
    const unsigned short* p = baseT + (size_t)(n0 + c) * row_stride + k0 + half * 8;
    ABFrag f;
    uint4 a = *(const uint4*)(p);
    uint4 b = *(const uint4*)(p + 16);
    f.u[0] = a.x; f.u[1] = a.y; f.u[2] = a.z; f.u[3] = a.w;
    f.u[4] = b.x; f.u[5] = b.y; f.u[6] = b.z; f.u[7] = b.w;
    return f.v;
}

static __device__ __forceinline__ v8f wmma_bf16(v16bf a, v16bf b, v8f c) {
    return __builtin_amdgcn_wmma_f32_16x16x32_bf16(false, a, false, b, (short)0, c, false, false);
}

// ---------------------------------------------------------------------------
// Problem constants
// ---------------------------------------------------------------------------
#define BB    4096
#define NN    200
#define NPAD  208     // 13 * 16  (M padding)
#define KPAD  224     // 7  * 32  (K padding for a@x)
#define FIN   64
#define CC    32
#define NC    6400    // N * C
#define H1D   512
#define H2D   256
#define H3D   128

// LDS layout for cheb kernel (ushort element offsets)
#define XB_OFF   0                       // [208][64]
#define XBT_OFF  (XB_OFF + NPAD*FIN)     // [64][224]
#define T1_OFF   (XBT_OFF + FIN*KPAD)    // [208][64]
#define W0T_OFF  (T1_OFF + NPAD*FIN)     // [32][64]
#define W1T_OFF  (W0T_OFF + CC*FIN)      // [32][64]
#define BC_OFF   (W1T_OFF + CC*FIN)      // 32 floats (64 ushort slots)
#define CHEB_LDS_USHORTS (BC_OFF + 64)
#define CHEB_LDS_BYTES   (CHEB_LDS_USHORTS * 2)

// ---------------------------------------------------------------------------
// Pre-pass: f32 -> bf16 conversions
// ---------------------------------------------------------------------------
__global__ void convert_bf16_kernel(const float* __restrict__ in,
                                    unsigned short* __restrict__ out, int n) {
    int i = blockIdx.x * 256 + threadIdx.x;
    if (i < n) out[i] = f2bf(in[i]);
}

// a [200,200] f32 -> zero-padded bf16 [208,224]
__global__ void convert_a_kernel(const float* __restrict__ a,
                                 unsigned short* __restrict__ out) {
    int i = blockIdx.x * 256 + threadIdx.x;
    if (i >= NPAD * KPAD) return;
    int row = i / KPAD, col = i - row * KPAD;
    float v = (row < NN && col < NN) ? a[row * NN + col] : 0.0f;
    out[i] = f2bf(v);
}

// ---------------------------------------------------------------------------
// ChebConv (K=2) + ELU fused kernel: one workgroup (8 waves) per batch.
//   T1 = A_pad @ Xb            (M=208, K=224, N=64)   -> LDS
//   Hb = Xb@W0 + T1@W1 + b     (M=208, K=64,  N=32)   -> ELU -> global bf16
// ---------------------------------------------------------------------------
__global__ void __launch_bounds__(256)
cheb_kernel(const float* __restrict__ x,
            const float* __restrict__ w0,
            const float* __restrict__ w1,
            const float* __restrict__ bconv,
            const unsigned short* __restrict__ a_bf,   // [208][224]
            unsigned short* __restrict__ Hout) {       // [B][6400]
    extern __shared__ unsigned short smem[];
    unsigned short* Xb  = smem + XB_OFF;    // [208][64]
    unsigned short* XbT = smem + XBT_OFF;   // [64][224]
    unsigned short* T1  = smem + T1_OFF;    // [208][64]
    unsigned short* W0t = smem + W0T_OFF;   // [32][64]
    unsigned short* W1t = smem + W1T_OFF;   // [32][64]
    float*          bcl = (float*)(smem + BC_OFF);

    const int tid  = threadIdx.x;
    const int lane = tid & 31;
    const int wave = tid >> 5;
    const int b    = blockIdx.x;

    // ---- zero Xb + XbT padding regions (they are contiguous: 27648 ushorts) ----
    {
        unsigned int* z = (unsigned int*)smem;
        for (int i = tid; i < (NPAD * FIN + FIN * KPAD) / 2; i += 256) z[i] = 0u;
    }
    __syncthreads();

    // ---- stage x[b] (f32) -> bf16 into Xb (row major) and XbT (transposed) ----
    {
        const float4* src = (const float4*)(x + (size_t)b * NN * FIN);
        for (int i = tid; i < (NN * FIN) / 4; i += 256) {  // 3200 float4
            float4 v = src[i];
            int n  = i >> 4;           // (i*4)/64
            int f0 = (i & 15) * 4;
            unsigned short h0 = f2bf(v.x), h1 = f2bf(v.y), h2 = f2bf(v.z), h3 = f2bf(v.w);
            ((uint2*)(Xb + n * FIN + f0))[0] = make_uint2(pack2(h0, h1), pack2(h2, h3));
            XbT[(f0 + 0) * KPAD + n] = h0;
            XbT[(f0 + 1) * KPAD + n] = h1;
            XbT[(f0 + 2) * KPAD + n] = h2;
            XbT[(f0 + 3) * KPAD + n] = h3;
        }
        for (int i = tid; i < FIN * CC; i += 256) {        // [f][c] -> Wt[c][f]
            int f = i >> 5, c = i & 31;
            W0t[c * FIN + f] = f2bf(w0[i]);
            W1t[c * FIN + f] = f2bf(w1[i]);
        }
        if (tid < CC) bcl[tid] = bconv[tid];
    }
    __syncthreads();

    // ---- T1 = A_pad @ Xb : 13 x 4 tiles of 16x16, K = 7 chunks of 32 ----
    for (int t = wave; t < 13 * 4; t += 8) {
        const int mt = t % 13, nt = t / 13;
        v8f acc = {};
        #pragma unroll
        for (int kc = 0; kc < 7; ++kc) {
            v16bf af = load_a_frag(a_bf, KPAD, mt * 16, kc * 32, lane);  // global (L2-resident)
            v16bf bf = load_b_frag(XbT, KPAD, nt * 16, kc * 32, lane);   // LDS
            acc = wmma_bf16(af, bf, acc);
        }
        const int c  = nt * 16 + (lane & 15);
        const int m0 = mt * 16 + (lane >> 4) * 8;
        #pragma unroll
        for (int r = 0; r < 8; ++r) T1[(m0 + r) * FIN + c] = f2bf(acc[r]);
    }
    __syncthreads();

    // ---- Hb = Xb@W0 + T1@W1 + b, ELU : 13 x 2 tiles, K = 2 chunks of 32 ----
    for (int t = wave; t < 13 * 2; t += 8) {
        const int mt = t % 13, ct = t / 13;
        v8f acc = {};
        #pragma unroll
        for (int kc = 0; kc < 2; ++kc) {
            acc = wmma_bf16(load_a_frag(Xb, FIN, mt * 16, kc * 32, lane),
                            load_b_frag(W0t, FIN, ct * 16, kc * 32, lane), acc);
            acc = wmma_bf16(load_a_frag(T1, FIN, mt * 16, kc * 32, lane),
                            load_b_frag(W1t, FIN, ct * 16, kc * 32, lane), acc);
        }
        const int c    = ct * 16 + (lane & 15);
        const float bc = bcl[c];
        const int m0   = mt * 16 + (lane >> 4) * 8;
        #pragma unroll
        for (int r = 0; r < 8; ++r) {
            int m = m0 + r;
            if (m < NN) {
                float v = acc[r] + bc;
                v = (v > 0.0f) ? v : (__expf(v) - 1.0f);  // ELU
                Hout[(size_t)b * NC + m * CC + c] = f2bf(v);
            }
        }
    }
}

// ---------------------------------------------------------------------------
// Generic bf16 WMMA GEMM + bias + ReLU:  C[M,N] = relu(A[M,K] @ B[K,N] + bias)
// Block tile 128x64, 256 threads = 8 waves in 4x2, each wave 2x2 16x16 tiles.
// Double-buffered: A tile staged via global_load_async_to_lds_b128 (ASYNCcnt),
// B tile staged synchronously (needs transpose).  K%32==0, M%128==0, N%64==0.
// ---------------------------------------------------------------------------
__global__ void __launch_bounds__(256)
gemm_bias_relu_kernel(const unsigned short* __restrict__ A,
                      const unsigned short* __restrict__ B,
                      const float* __restrict__ bias,
                      unsigned short* __restrict__ C,
                      int M, int N, int K) {
    __shared__ unsigned short As[2][128 * 32];   // [128][32] x2
    __shared__ unsigned short Bt[2][64 * 32];    // transposed [n][k] x2

    const int tid  = threadIdx.x;
    const int lane = tid & 31;
    const int wave = tid >> 5;
    const int wm   = wave >> 1;               // 0..3
    const int wn   = wave & 1;                // 0..1
    const int m0   = blockIdx.x * 128;
    const int n0   = blockIdx.y * 64;

    v8f acc[2][2] = {{{}, {}}, {{}, {}}};

    // per-thread addressing for the tile loaders
    const int arow = tid >> 1;                // 0..127 (2 threads per row)
    const int ak8  = (tid & 1) * 16;          // 16 bf16 = 32B per thread in 2x b128
    const int bk   = tid >> 3;                // 0..31
    const int bn8  = (tid & 7) * 8;           // 8 bf16 per thread

    auto load_a_tile = [&](int kc, int buf) {
        const unsigned short* src = A + (size_t)(m0 + arow) * K + kc * 32 + ak8;
        unsigned short* dst = &As[buf][arow * 32 + ak8];
        async_copy_b128(src, dst);
        async_copy_b128(src + 8, dst + 8);
    };
    auto load_b_tile = [&](int kc, int buf) {
        uint4 v = *(const uint4*)(B + (size_t)(kc * 32 + bk) * N + n0 + bn8);
        const unsigned short* e = (const unsigned short*)&v;
        #pragma unroll
        for (int j = 0; j < 8; ++j) Bt[buf][(bn8 + j) * 32 + bk] = e[j];
    };

    const int kChunks = K >> 5;
    load_a_tile(0, 0);
    load_b_tile(0, 0);
    wait_async0();
    __syncthreads();

    for (int kc = 0; kc < kChunks; ++kc) {
        const int cur = kc & 1;
        if (kc + 1 < kChunks) {
            load_a_tile(kc + 1, cur ^ 1);     // async into other buffer
            load_b_tile(kc + 1, cur ^ 1);
        }
        v16bf af0 = load_a_frag(As[cur], 32, wm * 32 + 0,  0, lane);
        v16bf af1 = load_a_frag(As[cur], 32, wm * 32 + 16, 0, lane);
        v16bf bf0 = load_b_frag(Bt[cur], 32, wn * 32 + 0,  0, lane);
        v16bf bf1 = load_b_frag(Bt[cur], 32, wn * 32 + 16, 0, lane);
        acc[0][0] = wmma_bf16(af0, bf0, acc[0][0]);
        acc[0][1] = wmma_bf16(af0, bf1, acc[0][1]);
        acc[1][0] = wmma_bf16(af1, bf0, acc[1][0]);
        acc[1][1] = wmma_bf16(af1, bf1, acc[1][1]);
        wait_async0();
        __syncthreads();
    }

    #pragma unroll
    for (int mt = 0; mt < 2; ++mt) {
        #pragma unroll
        for (int nt = 0; nt < 2; ++nt) {
            const int cn = n0 + wn * 32 + nt * 16 + (lane & 15);
            const float bs = bias[cn];
            const int mr0 = m0 + wm * 32 + mt * 16 + (lane >> 4) * 8;
            #pragma unroll
            for (int r = 0; r < 8; ++r) {
                float v = acc[mt][nt][r] + bs;
                v = fmaxf(v, 0.0f);
                C[(size_t)(mr0 + r) * N + cn] = f2bf(v);
            }
        }
    }
}

// ---------------------------------------------------------------------------
// Head: out[b] = sigmoid(H3[b,:] . w4 + b4)   (128-wide dot per row)
// ---------------------------------------------------------------------------
__global__ void __launch_bounds__(256)
head_kernel(const unsigned short* __restrict__ H3,
            const float* __restrict__ w4,
            const float* __restrict__ b4,
            float* __restrict__ out) {
    int b = blockIdx.x * 256 + threadIdx.x;
    if (b >= BB) return;
    const unsigned short* row = H3 + (size_t)b * H3D;
    float s = 0.0f;
    #pragma unroll 4
    for (int i = 0; i < H3D; ++i) s += bf2f(row[i]) * w4[i];
    s += b4[0];
    out[b] = 1.0f / (1.0f + __expf(-s));
}

// ---------------------------------------------------------------------------
// Launch
// ---------------------------------------------------------------------------
extern "C" void kernel_launch(void* const* d_in, const int* in_sizes, int n_in,
                              void* d_out, int out_size, void* d_ws, size_t ws_size,
                              hipStream_t stream) {
    const float* x      = (const float*)d_in[0];
    const float* a      = (const float*)d_in[1];
    const float* w0     = (const float*)d_in[2];
    const float* w1     = (const float*)d_in[3];
    const float* b_conv = (const float*)d_in[4];
    const float* w_d1   = (const float*)d_in[5];
    const float* b_d1   = (const float*)d_in[6];
    const float* w_d2   = (const float*)d_in[7];
    const float* b_d2   = (const float*)d_in[8];
    const float* w_d3   = (const float*)d_in[9];
    const float* b_d3   = (const float*)d_in[10];
    const float* w_d4   = (const float*)d_in[11];
    const float* b_d4   = (const float*)d_in[12];
    float* out          = (float*)d_out;

    // workspace carving (256B aligned)
    char* ws = (char*)d_ws;
    size_t off = 0;
    auto carve = [&](size_t bytes) {
        char* p = ws + off;
        off = (off + bytes + 255) & ~(size_t)255;
        return p;
    };
    unsigned short* a_bf  = (unsigned short*)carve((size_t)NPAD * KPAD * 2);
    unsigned short* wd1bf = (unsigned short*)carve((size_t)NC * H1D * 2);
    unsigned short* wd2bf = (unsigned short*)carve((size_t)H1D * H2D * 2);
    unsigned short* wd3bf = (unsigned short*)carve((size_t)H2D * H3D * 2);
    unsigned short* Hws   = (unsigned short*)carve((size_t)BB * NC * 2);
    unsigned short* H1ws  = (unsigned short*)carve((size_t)BB * H1D * 2);
    unsigned short* H2ws  = (unsigned short*)carve((size_t)BB * H2D * 2);
    unsigned short* H3ws  = (unsigned short*)carve((size_t)BB * H3D * 2);
    (void)ws_size; (void)in_sizes; (void)n_in; (void)out_size;

    // pre-pass conversions
    convert_a_kernel<<<(NPAD * KPAD + 255) / 256, 256, 0, stream>>>(a, a_bf);
    convert_bf16_kernel<<<(NC * H1D + 255) / 256, 256, 0, stream>>>(w_d1, wd1bf, NC * H1D);
    convert_bf16_kernel<<<(H1D * H2D + 255) / 256, 256, 0, stream>>>(w_d2, wd2bf, H1D * H2D);
    convert_bf16_kernel<<<(H2D * H3D + 255) / 256, 256, 0, stream>>>(w_d3, wd3bf, H2D * H3D);

    // ChebConv + ELU (dynamic LDS ~90KB; WGP has 320KB)
    (void)hipFuncSetAttribute((const void*)cheb_kernel,
                              hipFuncAttributeMaxDynamicSharedMemorySize,
                              CHEB_LDS_BYTES);
    cheb_kernel<<<BB, 256, CHEB_LDS_BYTES, stream>>>(x, w0, w1, b_conv, a_bf, Hws);

    // dense stack
    gemm_bias_relu_kernel<<<dim3(BB / 128, H1D / 64), 256, 0, stream>>>(
        Hws, wd1bf, b_d1, H1ws, BB, H1D, NC);
    gemm_bias_relu_kernel<<<dim3(BB / 128, H2D / 64), 256, 0, stream>>>(
        H1ws, wd2bf, b_d2, H2ws, BB, H2D, H1D);
    gemm_bias_relu_kernel<<<dim3(BB / 128, H3D / 64), 256, 0, stream>>>(
        H2ws, wd3bf, b_d3, H3ws, BB, H3D, H2D);

    // head
    head_kernel<<<BB / 256, 256, 0, stream>>>(H3ws, w_d4, b_d4, out);
}
